// StandardAttention_12120397709584
// MI455X (gfx1250) — matrix operand order repaired and checked by reading, not compiled, not run on hardware
//
#include <hip/hip_runtime.h>
#include <cstdint>
#include <cstddef>

// ---------------------------------------------------------------------------
// CDNA5 (gfx1250, wave32) WMMA transformer block with async-LDS pipelining.
//  D = A(16x32 f16) x B(32x16 f16) + C(16x16 f32) via v_wmma_f32_16x16x32_f16
//  Fragment layouts per CDNA5 ISA 7.12.2:
//   A: lane m=lane&15, khalf=lane>>4; v16h elems 0..7 -> K=8*khalf..+7,
//      elems 8..15 -> K=16+8*khalf..+7  (two b128 reads from row-major [m][k])
//   B: lane n=lane&15, khalf=lane>>4; same K order as A (elems 0..7 -> K=8*khalf..+7,
//      elems 8..15 -> K=16+8*khalf..+7; two b128 reads from B transposed [n][k])
//   C/D: lane n=lane&15; reg r -> m = r + 8*(lane>>4)
//  Tile staging: GLOBAL_LOAD_ASYNC_TO_LDS_B128 (ASYNCcnt) when available,
//  double-buffered so the next tile streams in while WMMAs run.
// ---------------------------------------------------------------------------

typedef __attribute__((ext_vector_type(16))) _Float16 v16h;
typedef __attribute__((ext_vector_type(8)))  float    v8f;
typedef __attribute__((ext_vector_type(4)))  int      v4i;

union Frag16 { v16h h; uint4 u[2]; };

__device__ __forceinline__ v8f wmma16(const v16h a, const v16h b, const v8f c) {
  return __builtin_amdgcn_wmma_f32_16x16x32_f16(false, a, false, b,
                                                (short)0, c, false, false);
}

#if defined(__gfx1250__) && __has_builtin(__builtin_amdgcn_global_load_async_to_lds_b128)
#define HAVE_ASYNC_LDS 1
#endif

// 16-byte global -> LDS transfer (async when supported, sync fallback)
__device__ __forceinline__ void ld16_to_lds(void* lds, const void* g) {
#ifdef HAVE_ASYNC_LDS
  __builtin_amdgcn_global_load_async_to_lds_b128(
      (__attribute__((address_space(1))) v4i*)(void*)g,
      (__attribute__((address_space(3))) v4i*)lds, 0, 0);
#else
  *(uint4*)lds = *(const uint4*)g;
#endif
}

__device__ __forceinline__ void async_wait0() {
#ifdef HAVE_ASYNC_LDS
#if __has_builtin(__builtin_amdgcn_s_wait_asynccnt)
  __builtin_amdgcn_s_wait_asynccnt(0);
#else
  asm volatile("s_wait_asynccnt 0" ::: "memory");
#endif
#endif
}

// ------------------------- weight convert + transpose -----------------------
// wt[n*K + k] = (f16) w[k*N + n]   (writes coalesced; one-shot preprocessing)
__global__ __launch_bounds__(256) void k_conv_transpose(
    const float* __restrict__ w, _Float16* __restrict__ wt,
    unsigned K, unsigned N) {
  unsigned idx = blockIdx.x * 256u + threadIdx.x;
  if (idx >= K * N) return;
  unsigned n = idx / K, k = idx - n * K;
  wt[idx] = (_Float16)w[(size_t)k * N + n];
}

// ------------------------------- layernorm ----------------------------------
// one block per row of D=1024; f32 stats, f16 output
__global__ __launch_bounds__(256) void k_layernorm(
    const float* __restrict__ x, const float* __restrict__ g,
    const float* __restrict__ b, _Float16* __restrict__ out) {
  constexpr int D = 1024;
  __shared__ float red[16];
  const int row = blockIdx.x;
  const int tid = threadIdx.x;
  const float* xr = x + (size_t)row * D;
  float4 v = *(const float4*)(xr + tid * 4);
  float s  = v.x + v.y + v.z + v.w;
  float s2 = v.x * v.x + v.y * v.y + v.z * v.z + v.w * v.w;
#pragma unroll
  for (int off = 1; off < 32; off <<= 1) {
    s  += __shfl_xor(s,  off);
    s2 += __shfl_xor(s2, off);
  }
  if ((tid & 31) == 0) { red[(tid >> 5) * 2] = s; red[(tid >> 5) * 2 + 1] = s2; }
  __syncthreads();
  float ts = 0.f, ts2 = 0.f;
#pragma unroll
  for (int wv = 0; wv < 8; ++wv) { ts += red[wv * 2]; ts2 += red[wv * 2 + 1]; }
  const float mu   = ts / D;
  const float var  = ts2 / D - mu * mu;
  const float rstd = rsqrtf(var + 1e-5f);
  const int d0 = tid * 4;
  _Float16* o = out + (size_t)row * D + d0;
  o[0] = (_Float16)((v.x - mu) * rstd * g[d0 + 0] + b[d0 + 0]);
  o[1] = (_Float16)((v.y - mu) * rstd * g[d0 + 1] + b[d0 + 1]);
  o[2] = (_Float16)((v.z - mu) * rstd * g[d0 + 2] + b[d0 + 2]);
  o[3] = (_Float16)((v.w - mu) * rstd * g[d0 + 3] + b[d0 + 3]);
}

// ------------------------------ WMMA GEMM -----------------------------------
// C[M,N] = epilogue( A[M,K] @ Bt[N,K]^T + bias )
// block tile 128x128, BK=32; 8 waves, each wave 32x64 (2x4 WMMA tiles).
// Double-buffered LDS; async prefetch of tile t+1 overlaps WMMAs of tile t.
// MODE 0: scatter q/k/v (v transposed), 1: +resid -> f32, 2: GELU -> f16,
// MODE 3: +resid -> f32 (final output)
template <int MODE>
__global__ __launch_bounds__(256) void k_gemm_wmma(
    const _Float16* __restrict__ A, const _Float16* __restrict__ Bt,
    const float* __restrict__ bias, const float* __restrict__ resid,
    float* __restrict__ outF, _Float16* __restrict__ outH,
    _Float16* __restrict__ qo, _Float16* __restrict__ ko,
    _Float16* __restrict__ vo, int M, int N, int K) {
  constexpr int BM = 128, BN = 128, BK = 32, LD = BK + 8;
  __shared__ _Float16 As[2][BM][LD];
  __shared__ _Float16 Bs[2][BN][LD];

  const int tid  = threadIdx.x;
  const int lane = tid & 31;
  const int wave = tid >> 5;
  const int wm   = wave & 3;   // 4 waves along M (32 rows each)
  const int wn   = wave >> 2;  // 2 waves along N (64 cols each)
  const int l15  = lane & 15;
  const int grp  = lane >> 4;
  const int m0   = blockIdx.y * BM;
  const int n0   = blockIdx.x * BN;

  v8f acc[2][4];
#pragma unroll
  for (int mt = 0; mt < 2; ++mt)
#pragma unroll
    for (int nt = 0; nt < 4; ++nt) acc[mt][nt] = v8f{};

  auto stage = [&](int buf, int kb) {
#pragma unroll
    for (int i = tid; i < BM * 4; i += 256) {
      const int row = i >> 2, seg = i & 3;
      ld16_to_lds(&As[buf][row][seg * 8],
                  &A[(size_t)(m0 + row) * K + kb + seg * 8]);
    }
#pragma unroll
    for (int i = tid; i < BN * 4; i += 256) {
      const int row = i >> 2, seg = i & 3;
      ld16_to_lds(&Bs[buf][row][seg * 8],
                  &Bt[(size_t)(n0 + row) * K + kb + seg * 8]);
    }
  };

  const int nk = K / BK;
  stage(0, 0);
  async_wait0();
  __syncthreads();

  for (int t = 0; t < nk; ++t) {
    const int cur = t & 1;
    if (t + 1 < nk) stage(cur ^ 1, (t + 1) * BK);  // prefetch next tile

    Frag16 af[2];
#pragma unroll
    for (int mt = 0; mt < 2; ++mt) {
      const int r = wm * 32 + mt * 16 + l15;
      af[mt].u[0] = *(const uint4*)&As[cur][r][8 * grp];
      af[mt].u[1] = *(const uint4*)&As[cur][r][16 + 8 * grp];
    }
#pragma unroll
    for (int nt = 0; nt < 4; ++nt) {
      Frag16 bf;
      const int r = wn * 64 + nt * 16 + l15;
      bf.u[0] = *(const uint4*)&Bs[cur][r][8 * grp];
      bf.u[1] = *(const uint4*)&Bs[cur][r][16 + 8 * grp];
#pragma unroll
      for (int mt = 0; mt < 2; ++mt)
        acc[mt][nt] = wmma16(af[mt].h, bf.h, acc[mt][nt]);
    }
    if (t + 1 < nk) {
      async_wait0();
      __syncthreads();
    }
  }

  // epilogue (D layout: m = r + 8*grp, n = l15)
#pragma unroll
  for (int mt = 0; mt < 2; ++mt)
#pragma unroll
    for (int nt = 0; nt < 4; ++nt)
#pragma unroll
      for (int r = 0; r < 8; ++r) {
        const int mg = m0 + wm * 32 + mt * 16 + r + 8 * grp;
        const int ng = n0 + wn * 64 + nt * 16 + l15;
        float val = acc[mt][nt][r] + bias[ng];
        if (MODE == 0) {
          // N = 3072: split q/k/v, H=16, DH=64, S=2048
          const int part = ng >> 10, d = ng & 1023;
          const int h = d >> 6, dh = d & 63;
          const int bb = mg >> 11, ss = mg & 2047;
          const size_t bh = (size_t)(bb * 16 + h);
          const _Float16 hv = (_Float16)val;
          if (part == 0)      qo[(bh * 2048 + ss) * 64 + dh] = hv;
          else if (part == 1) ko[(bh * 2048 + ss) * 64 + dh] = hv;
          else                vo[(bh * 64 + dh) * 2048 + ss] = hv;
        } else if (MODE == 1) {
          const size_t i = (size_t)mg * N + ng;
          outF[i] = val + resid[i];
        } else if (MODE == 2) {
          const float ge = 0.5f * val * (1.0f + erff(val * 0.70710678118f));
          outH[(size_t)mg * N + ng] = (_Float16)ge;
        } else {  // MODE 3
          const size_t i = (size_t)mg * N + ng;
          outF[i] = val + resid[i];
        }
      }
}

// --------------------------- flash attention --------------------------------
// grid = B*H*(S/128); 8 waves, each wave owns 16 query rows x DH=64.
// Streams 64-key tiles (double-buffered, async prefetch):
//   scores (8 WMMA) -> online softmax -> P@V (8 WMMA).
__global__ __launch_bounds__(256) void k_flash_attn(
    const _Float16* __restrict__ Q, const _Float16* __restrict__ Kb,
    const _Float16* __restrict__ Vt, _Float16* __restrict__ Out) {
  constexpr int S = 2048, DH = 64;
  constexpr int LK = DH + 8;  // 72 halfs
  constexpr int LP = 64 + 8;  // 72 halfs
  __shared__ _Float16 kt[2][64][LK];    // [key][dh]
  __shared__ _Float16 vs[2][64][LP];    // [dh][key]  (V transposed)
  __shared__ _Float16 pb[8][16][LP];    // per-wave P transpose buffer

  const int blk  = blockIdx.x;
  const int qblk = blk & 15;       // S/128 = 16
  const int bh   = blk >> 4;       // 0..31
  const int tid  = threadIdx.x;
  const int lane = tid & 31;
  const int wave = tid >> 5;
  const int l15  = lane & 15;
  const int grp  = lane >> 4;
  const size_t baseQ = (size_t)bh * S * DH;
  const size_t baseV = (size_t)bh * DH * S;
  const int qrow0 = qblk * 128 + wave * 16;

  // preload Q fragments (A layout, 2 K-steps over DH=64)
  Frag16 qf[2];
  const _Float16* qp = Q + baseQ + (size_t)(qrow0 + l15) * DH;
  qf[0].u[0] = *(const uint4*)(qp + 8 * grp);
  qf[0].u[1] = *(const uint4*)(qp + 16 + 8 * grp);
  qf[1].u[0] = *(const uint4*)(qp + 32 + 8 * grp);
  qf[1].u[1] = *(const uint4*)(qp + 48 + 8 * grp);

  v8f oacc[4];
#pragma unroll
  for (int nt = 0; nt < 4; ++nt) oacc[nt] = v8f{};
  float mst[8], lst[8];
#pragma unroll
  for (int r = 0; r < 8; ++r) { mst[r] = -1e30f; lst[r] = 0.f; }

  auto stage = [&](int buf, int kb0) {
#pragma unroll
    for (int i = tid; i < 64 * 8; i += 256) {  // K tile [64][64]
      const int row = i >> 3, seg = i & 7;
      ld16_to_lds(&kt[buf][row][seg * 8],
                  &Kb[baseQ + (size_t)(kb0 + row) * DH + seg * 8]);
    }
#pragma unroll
    for (int i = tid; i < 64 * 8; i += 256) {  // V^T tile [64][64]
      const int row = i >> 3, seg = i & 7;
      ld16_to_lds(&vs[buf][row][seg * 8],
                  &Vt[baseV + (size_t)row * S + kb0 + seg * 8]);
    }
  };

  constexpr int NT = S / 64;  // 32 key tiles
  stage(0, 0);
  async_wait0();
  __syncthreads();

  for (int t = 0; t < NT; ++t) {
    const int cur = t & 1;
    if (t + 1 < NT) stage(cur ^ 1, (t + 1) * 64);  // prefetch next tile

    // scores: 16 q x 64 keys (K-dim = DH = 64 -> 2 WMMA per 16-key tile)
    v8f sacc[4];
#pragma unroll
    for (int nt = 0; nt < 4; ++nt) {
      sacc[nt] = v8f{};
      Frag16 kf0, kf1;
      const int r = nt * 16 + l15;  // key row in tile
      kf0.u[0] = *(const uint4*)&kt[cur][r][8 * grp];
      kf0.u[1] = *(const uint4*)&kt[cur][r][16 + 8 * grp];
      kf1.u[0] = *(const uint4*)&kt[cur][r][32 + 8 * grp];
      kf1.u[1] = *(const uint4*)&kt[cur][r][48 + 8 * grp];
      sacc[nt] = wmma16(qf[0].h, kf0.h, sacc[nt]);
      sacc[nt] = wmma16(qf[1].h, kf1.h, sacc[nt]);
    }

    // online softmax (row reductions across the 16-lane group)
    const float scale = 0.125f;  // 1/sqrt(64)
#pragma unroll
    for (int r = 0; r < 8; ++r) {
      float mtile = -1e30f;
#pragma unroll
      for (int nt = 0; nt < 4; ++nt) {
        const float sv = sacc[nt][r] * scale;
        sacc[nt][r] = sv;
        mtile = fmaxf(mtile, sv);
      }
#pragma unroll
      for (int off = 1; off < 16; off <<= 1)
        mtile = fmaxf(mtile, __shfl_xor(mtile, off));
      const float mnew  = fmaxf(mst[r], mtile);
      const float alpha = __expf(mst[r] - mnew);
      float rs = 0.f;
#pragma unroll
      for (int nt = 0; nt < 4; ++nt) {
        const float p = __expf(sacc[nt][r] - mnew);
        pb[wave][r + 8 * grp][nt * 16 + l15] = (_Float16)p;
        rs += p;
      }
#pragma unroll
      for (int off = 1; off < 16; off <<= 1) rs += __shfl_xor(rs, off);
      lst[r] = lst[r] * alpha + rs;
      mst[r] = mnew;
#pragma unroll
      for (int nt = 0; nt < 4; ++nt) oacc[nt][r] *= alpha;
    }

    // wave-internal LDS ordering before cross-lane re-read of P
    asm volatile("s_wait_dscnt 0" ::: "memory");

    // P fragments (A layout over keys)
    Frag16 pf[2];
    const _Float16* prow = &pb[wave][l15][0];
    pf[0].u[0] = *(const uint4*)(prow + 8 * grp);
    pf[0].u[1] = *(const uint4*)(prow + 16 + 8 * grp);
    pf[1].u[0] = *(const uint4*)(prow + 32 + 8 * grp);
    pf[1].u[1] = *(const uint4*)(prow + 48 + 8 * grp);

#pragma unroll
    for (int nt = 0; nt < 4; ++nt) {
      Frag16 vf0, vf1;
      const int r = nt * 16 + l15;  // dh row
      vf0.u[0] = *(const uint4*)&vs[cur][r][8 * grp];
      vf0.u[1] = *(const uint4*)&vs[cur][r][16 + 8 * grp];
      vf1.u[0] = *(const uint4*)&vs[cur][r][32 + 8 * grp];
      vf1.u[1] = *(const uint4*)&vs[cur][r][48 + 8 * grp];
      oacc[nt] = wmma16(pf[0].h, vf0.h, oacc[nt]);
      oacc[nt] = wmma16(pf[1].h, vf1.h, oacc[nt]);
    }

    if (t + 1 < NT) {
      async_wait0();
      __syncthreads();
    }
  }

  // finalize: O /= l, write [B,S,H*DH] f16 (contiguous for out-proj GEMM)
  const int bb = bh >> 4, hh = bh & 15;
#pragma unroll
  for (int nt = 0; nt < 4; ++nt)
#pragma unroll
    for (int r = 0; r < 8; ++r) {
      const int mq = qrow0 + r + 8 * grp;
      const int dh = nt * 16 + l15;
      const float o = oacc[nt][r] / lst[r];
      Out[((size_t)(bb * 2048 + mq)) * 1024 + hh * 64 + dh] = (_Float16)o;
    }
}

// ----------------------------------------------------------------------------
extern "C" void kernel_launch(void* const* d_in, const int* in_sizes, int n_in,
                              void* d_out, int out_size, void* d_ws,
                              size_t ws_size, hipStream_t stream) {
  const float* x     = (const float*)d_in[0];
  const float* ln1_g = (const float*)d_in[1];
  const float* ln1_b = (const float*)d_in[2];
  const float* w_qkv = (const float*)d_in[3];
  const float* b_qkv = (const float*)d_in[4];
  const float* w_out = (const float*)d_in[5];
  const float* b_out = (const float*)d_in[6];
  const float* ln2_g = (const float*)d_in[7];
  const float* ln2_b = (const float*)d_in[8];
  const float* w1    = (const float*)d_in[9];
  const float* b1    = (const float*)d_in[10];
  const float* w2    = (const float*)d_in[11];
  const float* b2    = (const float*)d_in[12];

  constexpr size_t BS = 4096;   // B*S
  constexpr size_t D  = 1024;
  char* p = (char*)d_ws;
  auto take = [&](size_t bytes) { char* r = p; p += bytes; return r; };
  _Float16* wqkvT = (_Float16*)take(3072ull * 1024 * 2);
  _Float16* woutT = (_Float16*)take(1024ull * 1024 * 2);
  _Float16* w1T   = (_Float16*)take(4096ull * 1024 * 2);
  _Float16* w2T   = (_Float16*)take(1024ull * 4096 * 2);
  _Float16* h1    = (_Float16*)take(BS * D * 2);
  _Float16* h2    = (_Float16*)take(BS * D * 2);
  float*    x2    = (float*)   take(BS * D * 4);
  char* ub = take(4ull * BS * D * 2);          // 32 MB union region
  _Float16* qbuf  = (_Float16*)ub;             // [B,H,S,DH]
  _Float16* kbuf  = qbuf + BS * D;             // [B,H,S,DH]
  _Float16* vbufT = kbuf + BS * D;             // [B,H,DH,S]
  _Float16* attn  = vbufT + BS * D;            // [B,S,D]
  _Float16* ffh   = (_Float16*)ub;             // [BS,4096] aliases q..attn

  // 1) weights -> f16, transposed [N][K]
  k_conv_transpose<<<(3072u * 1024u + 255) / 256, 256, 0, stream>>>(w_qkv, wqkvT, 1024, 3072);
  k_conv_transpose<<<(1024u * 1024u + 255) / 256, 256, 0, stream>>>(w_out, woutT, 1024, 1024);
  k_conv_transpose<<<(4096u * 1024u + 255) / 256, 256, 0, stream>>>(w1, w1T, 1024, 4096);
  k_conv_transpose<<<(4096u * 1024u + 255) / 256, 256, 0, stream>>>(w2, w2T, 4096, 1024);

  // 2) LN1
  k_layernorm<<<BS, 256, 0, stream>>>(x, ln1_g, ln1_b, h1);

  // 3) QKV GEMM [4096,1024]x[1024,3072] -> q/k/v scatter
  k_gemm_wmma<0><<<dim3(3072 / 128, 4096 / 128), 256, 0, stream>>>(
      h1, wqkvT, b_qkv, nullptr, nullptr, nullptr, qbuf, kbuf, vbufT,
      4096, 3072, 1024);

  // 4) flash attention
  k_flash_attn<<<2 * 16 * (2048 / 128), 256, 0, stream>>>(qbuf, kbuf, vbufT, attn);

  // 5) out-proj + residual -> x2 (f32)
  k_gemm_wmma<1><<<dim3(1024 / 128, 4096 / 128), 256, 0, stream>>>(
      attn, woutT, b_out, x, x2, nullptr, nullptr, nullptr, nullptr,
      4096, 1024, 1024);

  // 6) LN2
  k_layernorm<<<BS, 256, 0, stream>>>(x2, ln2_g, ln2_b, h2);

  // 7) FFN1 + exact GELU -> ffh (f16)  [4096,1024]x[1024,4096]
  k_gemm_wmma<2><<<dim3(4096 / 128, 4096 / 128), 256, 0, stream>>>(
      h2, w1T, b1, nullptr, nullptr, ffh, nullptr, nullptr, nullptr,
      4096, 4096, 1024);

  // 8) FFN2 + residual -> d_out (f32)  [4096,4096]x[4096,1024]
  k_gemm_wmma<3><<<dim3(1024 / 128, 4096 / 128), 256, 0, stream>>>(
      ffh, w2T, b2, x2, (float*)d_out, nullptr, nullptr, nullptr, nullptr,
      4096, 1024, 4096);
}
